// EEG_LSTM_Model_63780264345743
// MI455X (gfx1250) — matrix operand order repaired and checked by reading, not compiled, not run on hardware
//
#include <hip/hip_runtime.h>

typedef _Float16 v16h __attribute__((ext_vector_type(16)));
typedef _Float16 v8h  __attribute__((ext_vector_type(8)));
typedef _Float16 v2h  __attribute__((ext_vector_type(2)));
typedef float    v8f  __attribute__((ext_vector_type(8)));
typedef float    v4f  __attribute__((ext_vector_type(4)));
typedef int      v8i  __attribute__((ext_vector_type(8)));

constexpr int kE = 19;
constexpr int kT = 2048;
constexpr int kH = 32;

__device__ __forceinline__ float fast_sigmoid(float x) {
  // sigmoid(x) = 1 / (1 + 2^(-x*log2(e)))
  float t = __builtin_amdgcn_exp2f(-1.4426950408889634f * x);
  return __builtin_amdgcn_rcpf(1.0f + t);
}

__device__ __forceinline__ float fast_tanh(float x) {
  x = fminf(fmaxf(x, -12.0f), 12.0f);
  float t = __builtin_amdgcn_exp2f(-2.8853900817779268f * x);  // e^(-2x)
  return (1.0f - t) * __builtin_amdgcn_rcpf(1.0f + t);
}

// One wave (32 lanes) per (electrode, direction, 16-batch tile): 19*2*4 = 152 waves.
// Orientation: D[gate,batch] = A(w_hh)[gate,K=h] x B(h^T)[K=h,batch] + C(x*w_ih + bias).
//  - lane%16 = batch column (for B/D), lane%16 = gate row (for A); lane/16 = K/M half split.
//  - h state lives per-lane as 8 packed-f16 dwords; D->B relayout for the next step is a
//    single SWAPX16 ds_swizzle exchange between lane halves (no LDS memory traffic).
__global__ __launch_bounds__(32) void eeg_lstm_wmma_kernel(
    const float* __restrict__ x,     // [B=64, T=2048, E=19]
    const float* __restrict__ w_ih,  // [E, 2, 128]
    const float* __restrict__ w_hh,  // [E, 2, 128, 32]
    const float* __restrict__ b_ih,  // [E, 2, 128]
    const float* __restrict__ b_hh,  // [E, 2, 128]
    float* __restrict__ hT)          // [B, E, 2, 32]  (workspace)
{
  const int wid = blockIdx.x;          // 0..151
  const int e   = wid >> 3;            // electrode
  const int d   = (wid >> 2) & 1;      // direction
  const int bt  = wid & 3;             // batch tile (16 batches)
  const int lane = threadIdx.x & 31;
  const int lh   = lane >> 4;          // lane half (K/M split)
  const int ln   = lane & 15;
  const int ed   = e * 2 + d;

  // ---- Preload A operands: w_hh as f16 in 16-bit A-matrix layout (16x32, M=gate,K=h) ----
  // A[gt] halves j=0..15 : K = j + 16*lh ; M = ln.
  v16h A[8];
#pragma unroll
  for (int gt = 0; gt < 8; ++gt) {
    const float* wr = w_hh + ((size_t)ed * 128 + 16 * gt + ln) * kH + 16 * lh;
    v4f a0 = *(const v4f*)(wr + 0);
    v4f a1 = *(const v4f*)(wr + 4);
    v4f a2 = *(const v4f*)(wr + 8);
    v4f a3 = *(const v4f*)(wr + 12);
#pragma unroll
    for (int j = 0; j < 4; ++j) {
      A[gt][j]      = (_Float16)a0[j];
      A[gt][4 + j]  = (_Float16)a1[j];
      A[gt][8 + j]  = (_Float16)a2[j];
      A[gt][12 + j] = (_Float16)a3[j];
    }
  }

  // ---- Preload w_ih and (b_ih+b_hh) in D-matrix layout, packed f16 (g = 16*gt + v + 8*lh) ----
  v8h wihp[8], biasp[8];
#pragma unroll
  for (int gt = 0; gt < 8; ++gt) {
    const size_t gbase = (size_t)ed * 128 + 16 * gt + 8 * lh;
    const float* wi = w_ih + gbase;
    const float* bi = b_ih + gbase;
    const float* bh = b_hh + gbase;
#pragma unroll
    for (int v = 0; v < 8; ++v) {
      wihp[gt][v]  = (_Float16)wi[v];
      biasp[gt][v] = (_Float16)(bi[v] + bh[v]);
    }
  }

  // ---- State: c (f32) and h (packed f16) : index h = 16*ct + v + 8*lh, batch = ln ----
  float c[2][8];
  int   hpk[2][4];  // hpk[ct][m] packs h at hidden {16ct+2m, 16ct+2m+1} + 8*lh
#pragma unroll
  for (int ct = 0; ct < 2; ++ct) {
#pragma unroll
    for (int v = 0; v < 8; ++v) c[ct][v] = 0.0f;
#pragma unroll
    for (int m = 0; m < 4; ++m) hpk[ct][m] = 0;
  }

  const int b = bt * 16 + ln;
  const float* xb = x + (size_t)b * kT * kE + e;

  for (int t = 0; t < kT; ++t) {
    const int te = d ? (kT - 1 - t) : t;
    const float xv = xb[(size_t)te * kE];
    {  // prefetch 16 steps ahead
      int tp = d ? (te >= 16 ? te - 16 : 0) : (te + 16 < kT ? te + 16 : kT - 1);
      __builtin_prefetch(xb + (size_t)tp * kE, 0, 1);
    }

    // ---- Build shared B operand (h^T, 32x16, K=h, N=batch) from packed h state ----
    // Lane L<16 needs K 0..15: own ct0 pack + other-half ct0 pack.
    // Lane L>=16 needs K 16..31: other-half ct1 pack + own ct1 pack.
    int sw0[4], sw1[4];
#pragma unroll
    for (int m = 0; m < 4; ++m) {
      sw0[m] = __builtin_amdgcn_ds_swizzle(hpk[0][m], 0x401F);  // SWAPX16
      sw1[m] = __builtin_amdgcn_ds_swizzle(hpk[1][m], 0x401F);
    }
    v8i bi_;
#pragma unroll
    for (int m = 0; m < 4; ++m) {
      bi_[m]     = (lh == 0) ? hpk[0][m] : sw1[m];
      bi_[m + 4] = (lh == 0) ? sw0[m]    : hpk[1][m];
    }
    const v16h Bop = __builtin_bit_cast(v16h, bi_);

    // ---- Two groups of 4 WMMAs (i,f,g,o for 16 hidden each) + fused elementwise ----
#pragma unroll
    for (int ct = 0; ct < 2; ++ct) {
      v8f c_i, c_f, c_g, c_o;
#pragma unroll
      for (int v = 0; v < 8; ++v) {
        c_i[v] = xv * (float)wihp[ct][v]     + (float)biasp[ct][v];
        c_f[v] = xv * (float)wihp[2 + ct][v] + (float)biasp[2 + ct][v];
        c_g[v] = xv * (float)wihp[4 + ct][v] + (float)biasp[4 + ct][v];
        c_o[v] = xv * (float)wihp[6 + ct][v] + (float)biasp[6 + ct][v];
      }
      v8f gi = __builtin_amdgcn_wmma_f32_16x16x32_f16(false, A[ct],     false, Bop, (short)0, c_i, false, false);
      v8f gf = __builtin_amdgcn_wmma_f32_16x16x32_f16(false, A[2 + ct], false, Bop, (short)0, c_f, false, false);
      v8f gg = __builtin_amdgcn_wmma_f32_16x16x32_f16(false, A[4 + ct], false, Bop, (short)0, c_g, false, false);
      v8f go = __builtin_amdgcn_wmma_f32_16x16x32_f16(false, A[6 + ct], false, Bop, (short)0, c_o, false, false);

      float hn[8];
#pragma unroll
      for (int v = 0; v < 8; ++v) {
        float ig = fast_sigmoid(gi[v]);
        float fg = fast_sigmoid(gf[v]);
        float gv = fast_tanh(gg[v]);
        float og = fast_sigmoid(go[v]);
        float cn = fg * c[ct][v] + ig * gv;
        c[ct][v] = cn;
        hn[v] = og * fast_tanh(cn);
      }
#pragma unroll
      for (int m = 0; m < 4; ++m) {
        auto p = __builtin_amdgcn_cvt_pkrtz(hn[2 * m], hn[2 * m + 1]);
        hpk[ct][m] = __builtin_bit_cast(int, p);
      }
    }
  }

  // ---- Write final hidden state: hT[b][e][d][hidden] ----
  float* hout = hT + (((size_t)b * kE + e) * 2 + d) * kH;
#pragma unroll
  for (int ct = 0; ct < 2; ++ct) {
#pragma unroll
    for (int m = 0; m < 4; ++m) {
      v2h p = __builtin_bit_cast(v2h, hpk[ct][m]);
      hout[16 * ct + 2 * m + 8 * lh]     = (float)p[0];
      hout[16 * ct + 2 * m + 1 + 8 * lh] = (float)p[1];
    }
  }
}

// LayerNorm over 64 features per (b,e), mean over electrodes, FC -> logit. One wave per batch.
__global__ __launch_bounds__(32) void eeg_head_kernel(
    const float* __restrict__ hT,       // [B, E, 64]
    const float* __restrict__ ln_gamma, // [E, 64]
    const float* __restrict__ ln_beta,  // [E, 64]
    const float* __restrict__ fc_w,     // [64]
    const float* __restrict__ fc_b,     // [1]
    float* __restrict__ out)            // [B]
{
  const int b = blockIdx.x;
  const int l = threadIdx.x;  // lane handles features l and l+32
  float p0 = 0.0f, p1 = 0.0f;
  for (int e = 0; e < kE; ++e) {
    const float* r = hT + ((size_t)b * kE + e) * 64;
    float v0 = r[l], v1 = r[l + 32];
    float s = v0 + v1;
#pragma unroll
    for (int o = 16; o > 0; o >>= 1) s += __shfl_xor(s, o, 32);
    const float mu = s * (1.0f / 64.0f);
    float d0 = v0 - mu, d1 = v1 - mu;
    float q = d0 * d0 + d1 * d1;
#pragma unroll
    for (int o = 16; o > 0; o >>= 1) q += __shfl_xor(q, o, 32);
    const float inv = __builtin_amdgcn_rsqf(q * (1.0f / 64.0f) + 1e-5f);
    p0 += d0 * inv * ln_gamma[e * 64 + l]      + ln_beta[e * 64 + l];
    p1 += d1 * inv * ln_gamma[e * 64 + l + 32] + ln_beta[e * 64 + l + 32];
  }
  p0 *= (1.0f / 19.0f);
  p1 *= (1.0f / 19.0f);
  float acc = p0 * fc_w[l] + p1 * fc_w[l + 32];
#pragma unroll
  for (int o = 16; o > 0; o >>= 1) acc += __shfl_xor(acc, o, 32);
  if (l == 0) out[b] = acc + fc_b[0];
}

extern "C" void kernel_launch(void* const* d_in, const int* in_sizes, int n_in,
                              void* d_out, int out_size, void* d_ws, size_t ws_size,
                              hipStream_t stream) {
  const float* x        = (const float*)d_in[0];
  const float* w_ih     = (const float*)d_in[1];
  const float* w_hh     = (const float*)d_in[2];
  const float* b_ih     = (const float*)d_in[3];
  const float* b_hh     = (const float*)d_in[4];
  const float* ln_gamma = (const float*)d_in[5];
  const float* ln_beta  = (const float*)d_in[6];
  const float* fc_w     = (const float*)d_in[7];
  const float* fc_b     = (const float*)d_in[8];
  float* hT = (float*)d_ws;  // [64,19,2,32] f32 = 311296 bytes

  eeg_lstm_wmma_kernel<<<dim3(19 * 2 * 4), dim3(32), 0, stream>>>(
      x, w_ih, w_hh, b_ih, b_hh, hT);
  eeg_head_kernel<<<dim3(64), dim3(32), 0, stream>>>(
      hT, ln_gamma, ln_beta, fc_w, fc_b, (float*)d_out);
}
